// ConvTransBlock_15564961481157
// MI455X (gfx1250) — matrix operand order repaired and checked by reading, not compiled, not run on hardware
//
#include <hip/hip_runtime.h>
#include <hip/hip_bf16.h>

// ---------------- problem constants ----------------
constexpr int CDIM  = 192;
constexpr int CHEADS= 6;
constexpr int CHD   = 32;            // head dim
constexpr int CWIN  = 7;
constexpr int CK    = 49;            // taps
constexpr int CB    = 4;
constexpr int CH    = 56;
constexpr int CW    = 56;
constexpr int CHW   = CH * CW;       // 3136
constexpr int CNPIX = CB * CHW;      // 12544
constexpr int CPAD  = 3;
constexpr float CEPS = 1e-5f;
constexpr int CBH   = CB * CHEADS;   // 24
constexpr int CKP   = 64;            // taps padded to 64 M rows

// ---------------- WMMA vector types ----------------
typedef __attribute__((ext_vector_type(16))) __bf16 v16bf;
typedef __attribute__((ext_vector_type(8)))  __bf16 v8bf;
typedef __attribute__((ext_vector_type(4)))  __bf16 v4bf;
typedef __attribute__((ext_vector_type(8)))  float  v8f;

__device__ __forceinline__ __bf16 f2bf(float f) {
  unsigned u = __builtin_bit_cast(unsigned, f);
  unsigned r = (u + 0x7FFFu + ((u >> 16) & 1u)) >> 16;
  unsigned short s = (unsigned short)r;
  return __builtin_bit_cast(__bf16, s);
}
__device__ __forceinline__ float bf2f(__bf16 b) {
  unsigned u = ((unsigned)__builtin_bit_cast(unsigned short, b)) << 16;
  return __builtin_bit_cast(float, u);
}

// ---------------- workspace layout ----------------
constexpr size_t A256(size_t x) { return (x + 255) & ~(size_t)255; }
constexpr size_t OFF_SCALE = 0;                                           // 192 f32
constexpr size_t OFF_SHIFT = A256(OFF_SCALE + CDIM * 4);                  // 192 f32
constexpr size_t OFF_XN    = A256(OFF_SHIFT + CDIM * 4);                  // xn bf16 [b][192][hw]
constexpr size_t OFF_WQKV  = A256(OFF_XN    + (size_t)CNPIX * CDIM * 2);  // 576x192 bf16
constexpr size_t OFF_WPROJ = A256(OFF_WQKV  + (size_t)3 * CDIM * CDIM * 2);
constexpr size_t OFF_WC1   = A256(OFF_WPROJ + (size_t)CDIM * CDIM * 2);
constexpr size_t OFF_WC2   = A256(OFF_WC1   + (size_t)4 * CDIM * CDIM * 2);
constexpr size_t OFF_WDM   = A256(OFF_WC2   + (size_t)4 * CDIM * CDIM * 2); // [64][49][64] bf16 (M-padded)
constexpr size_t OFF_QK    = A256(OFF_WDM   + (size_t)CKP * CK * 64 * 2);   // [bh][hw][64] bf16
constexpr size_t OFF_V     = A256(OFF_QK    + (size_t)CBH * CHW * 64 * 2);  // [bh][32][hw] bf16
constexpr size_t OFF_ATT   = A256(OFF_V     + (size_t)CBH * CHD * CHW * 2); // [bh][49][hw] f32
constexpr size_t OFF_AV    = A256(OFF_ATT   + (size_t)CBH * CK * CHW * 4);  // [b][192][hw] bf16
constexpr size_t OFF_X1F   = A256(OFF_AV    + (size_t)CNPIX * CDIM * 2);    // f32
constexpr size_t OFF_X1B   = A256(OFF_X1F   + (size_t)CNPIX * CDIM * 4);    // bf16
constexpr size_t OFF_H     = A256(OFF_X1B   + (size_t)CNPIX * CDIM * 2);    // [b][768][hw] bf16

// ---------------- weight conversion ----------------
__global__ void cvt_bf16_kernel(__bf16* __restrict__ dst, const float* __restrict__ src, int n) {
  int i = blockIdx.x * blockDim.x + threadIdx.x;
  if (i < n) dst[i] = f2bf(src[i]);
}

// dm_w [49][64][7][7] -> padded [64][tap][ci] bf16 (rows 49..63 zero => branchless A loads)
__global__ void dmw_transpose_kernel(__bf16* __restrict__ dst, const float* __restrict__ src) {
  int i = blockIdx.x * blockDim.x + threadIdx.x;
  if (i >= CKP * CK * 64) return;
  int o   = i / (64 * CK);
  int r   = i % (64 * CK);
  int tap = r / 64;
  int ci  = r % 64;
  float v = (o < CK) ? src[((size_t)o * 64 + ci) * CK + tap] : 0.f;
  dst[((size_t)o * CK + tap) * 64 + ci] = f2bf(v);
}

// ---------------- BN stats -> folded scale/shift ----------------
__global__ void bn_stats_kernel(const float* __restrict__ x, const float* __restrict__ gamma,
                                const float* __restrict__ beta, float* __restrict__ scale,
                                float* __restrict__ shift) {
  __shared__ float s1[256], s2[256];
  const int c = blockIdx.x;
  float sum = 0.f, sq = 0.f;
  for (int i = threadIdx.x; i < CNPIX; i += 256) {
    int b = i / CHW, hw = i % CHW;
    float v = x[((size_t)b * CDIM + c) * CHW + hw];
    sum += v; sq += v * v;
  }
  s1[threadIdx.x] = sum; s2[threadIdx.x] = sq;
  __syncthreads();
  for (int st = 128; st > 0; st >>= 1) {
    if (threadIdx.x < st) { s1[threadIdx.x] += s1[threadIdx.x + st]; s2[threadIdx.x] += s2[threadIdx.x + st]; }
    __syncthreads();
  }
  if (threadIdx.x == 0) {
    float mean = s1[0] / (float)CNPIX;
    float var  = s2[0] / (float)CNPIX - mean * mean;
    float r    = rsqrtf(var + CEPS);
    float sc   = gamma[c] * r;
    scale[c] = sc;
    shift[c] = beta[c] - mean * sc;
  }
}

__global__ void normalize_kernel(const float* __restrict__ x, const float* __restrict__ scale,
                                 const float* __restrict__ shift, __bf16* __restrict__ xn) {
  int i = blockIdx.x * blockDim.x + threadIdx.x;
  if (i >= CNPIX * CDIM) return;
  int c = (i / CHW) % CDIM;
  xn[i] = f2bf(x[i] * scale[c] + shift[c]);
}

// ---------------- generic WMMA GEMM: C[M,N] = W[M,K] * X[K,N] ----------------
// X layout: [b][k][hw], N = 12544 pixels. Block: 256 thr = 8 waves.
// Mblock=64 (4 subtiles) x Nblock=32 (2 subtiles); K-step 64 => 2 independent WMMA per stage.
// MODE 0 QKV   -> scatter q*scale|k into qk[bh][hw][64], v into v[bh][32][hw]
// MODE 1 PROJ  -> x1 = shortcut + acc + b : fp32 + bf16 streams
// MODE 2 C1    -> gelu(acc + b) -> bf16
// MODE 3 C2    -> out = x1 + acc + b -> fp32 d_out
template <int MODE>
__global__ void gemm_wmma_kernel(const __bf16* __restrict__ W, const __bf16* __restrict__ X,
                                 const float* __restrict__ bias, const float* __restrict__ fin,
                                 float* __restrict__ fout, __bf16* __restrict__ bout0,
                                 __bf16* __restrict__ bout1, int Ktot) {
  __shared__ __attribute__((aligned(32))) __bf16 sA[64 * 64];   // [m][k]  8KB
  __shared__ __attribute__((aligned(32))) __bf16 sB[32 * 64];   // [n][k]  4KB (transposed)
  const int tid  = threadIdx.x;
  const int lane = tid & 31;
  const int wave = tid >> 5;
  const int nb = blockIdx.x;                 // 32-pixel N block
  const int mb = blockIdx.y;                 // 64-row  M block
  const int p0 = nb * 32;
  const int b   = p0 / CHW;                  // 3136 % 32 == 0: no batch crossing
  const int hw0 = p0 % CHW;
  const int m0 = mb * 64;
  const int mi = wave & 3, ni = wave >> 2;

  const int row  = lane & 15;
  const int hsel = (lane < 16) ? 0 : 16;     // A: lanes 0-15 K{0..15}, 16-31 K{16..31}
  const int kb   = (lane < 16) ? 0 : 16;     // B: 16 contiguous K per lane

  v8f acc0 = {}, acc1 = {};                  // independent accumulators per K-half
  for (int k0 = 0; k0 < Ktot; k0 += 64) {
    // A tile 64m x 64k: 2 x v8bf per thread
#pragma unroll
    for (int j = 0; j < 2; ++j) {
      int idx = (tid + j * 256) * 8;
      int m = idx >> 6, k = idx & 63;
      *(v8bf*)&sA[m * 64 + k] = *(const v8bf*)&W[(size_t)(m0 + m) * Ktot + k0 + k];
    }
    // B tile 64k x 32n -> transposed [n][k]
    {
      int k  = tid >> 2;                     // 0..63
      int nn = (tid & 3) * 8;                // 0,8,16,24
      v8bf val = *(const v8bf*)&X[((size_t)b * Ktot + k0 + k) * CHW + hw0 + nn];
#pragma unroll
      for (int j = 0; j < 8; ++j) sB[(nn + j) * 64 + k] = val[j];
    }
    __syncthreads();
    union { v16bf v; v8bf h2[2]; } a0, a1, b0, b1;
    const __bf16* arow = &sA[(mi * 16 + row) * 64];
    a0.h2[0] = *(const v8bf*)&arow[hsel];
    a0.h2[1] = *(const v8bf*)&arow[8 + hsel];
    a1.h2[0] = *(const v8bf*)&arow[32 + hsel];
    a1.h2[1] = *(const v8bf*)&arow[40 + hsel];
    const __bf16* brow = &sB[(ni * 16 + row) * 64];
    b0.v = *(const v16bf*)&brow[kb];
    b1.v = *(const v16bf*)&brow[32 + kb];
    acc0 = __builtin_amdgcn_wmma_f32_16x16x32_bf16(false, a0.v, false, b0.v,
                                                   (short)0, acc0, false, false);
    acc1 = __builtin_amdgcn_wmma_f32_16x16x32_bf16(false, a1.v, false, b1.v,
                                                   (short)0, acc1, false, false);
    __syncthreads();
  }

  // epilogue: lane holds C rows M = r + (lane<16?0:8), col N = lane%16
  const int hw = hw0 + ni * 16 + row;
  const int mrow0 = m0 + mi * 16 + ((lane < 16) ? 0 : 8);
#pragma unroll
  for (int r = 0; r < 8; ++r) {
    int m = mrow0 + r;
    float val = acc0[r] + acc1[r] + bias[m];
    if (MODE == 0) {
      int sec = m / CDIM, rem = m % CDIM;
      int head = rem / CHD, c = rem % CHD;
      int bh = b * CHEADS + head;
      if (sec == 0)
        bout0[((size_t)bh * CHW + hw) * 64 + c] = f2bf(val * 0.17677669529663687f); // 1/sqrt(32)
      else if (sec == 1)
        bout0[((size_t)bh * CHW + hw) * 64 + 32 + c] = f2bf(val);
      else
        bout1[((size_t)bh * CHD + c) * CHW + hw] = f2bf(val);
    } else if (MODE == 1) {
      size_t o = ((size_t)b * CDIM + m) * CHW + hw;
      float v1 = val + fin[o];
      fout[o]  = v1;
      bout0[o] = f2bf(v1);
    } else if (MODE == 2) {
      float g = 0.5f * val * (1.0f + erff(val * 0.70710678118654752f));  // exact gelu
      bout0[((size_t)b * (4 * CDIM) + m) * CHW + hw] = f2bf(g);
    } else {
      size_t o = ((size_t)b * CDIM + m) * CHW + hw;
      fout[o] = val + fin[o];
    }
  }
}

// ---------------- attention conv as implicit GEMM ----------------
// attn[bh][o][h][w] = sum_{ci,kh,kw} dmw_t[o][kh*7+kw][ci] * qk[bh][h+kh-3][w+kw-3][ci]
// Block: 256 thr = 8 waves; 2 output rows per block (waves 0-3 -> h0, 4-7 -> h0+1),
// each wave one 16-row M tile of the 64-padded outputs. N tile = 16 pixels.
// Nested kh/kw loops (no div/mod), pointer-incremented A, 2 independent accumulators.
__global__ void attn_conv_kernel(const __bf16* __restrict__ qk, const __bf16* __restrict__ wdm,
                                 const float* __restrict__ dm_b, float* __restrict__ attn) {
  __shared__ __attribute__((aligned(32))) __bf16 sQ[8 * 22 * 64];  // [row][col][ch] 22.5KB
  const int w0 = blockIdx.x * 16;
  const int h0 = blockIdx.y * 2;
  const int bh = blockIdx.z;
  const int tid = threadIdx.x, lane = tid & 31, wave = tid >> 5;

  // cooperative halo load: 8 rows x 22 cols x 64 ch, zero-padded OOB
  for (int g = tid; g < 8 * 22 * 8; g += 256) {
    int ch8 = (g & 7) * 8;
    int pos = g >> 3;
    int hrow = pos / 22, col = pos % 22;
    int gh = h0 + hrow - CPAD;
    int gw = w0 + col - CPAD;
    v8bf val = {};
    if (gh >= 0 && gh < CH && gw >= 0 && gw < CW)
      val = *(const v8bf*)&qk[((size_t)bh * CHW + gh * CW + gw) * 64 + ch8];
    *(v8bf*)&sQ[pos * 64 + ch8] = val;
  }
  __syncthreads();

  const int hsub = wave >> 2;            // which output row
  const int o0   = (wave & 3) * 16;      // M tile
  const int row  = lane & 15;
  const int hsel = (lane < 16) ? 0 : 16;
  const int kb   = (lane < 16) ? 0 : 16;
  const __bf16* ap = &wdm[(size_t)(o0 + row) * CK * 64];   // advances 64/tap
  v8f acc0 = {}, acc1 = {};
  for (int kh = 0; kh < CWIN; ++kh) {
    const __bf16* bprow = &sQ[(((kh + hsub) * 22) + row) * 64];  // + kw*64 per tap
#pragma unroll
    for (int kw = 0; kw < CWIN; ++kw) {
      union { v16bf v; v8bf h2[2]; } a0, a1, bm0, bm1;
      a0.h2[0] = *(const v8bf*)&ap[hsel];                  // padded M rows are zeros
      a0.h2[1] = *(const v8bf*)&ap[8 + hsel];
      a1.h2[0] = *(const v8bf*)&ap[32 + hsel];
      a1.h2[1] = *(const v8bf*)&ap[40 + hsel];
      const __bf16* bp = bprow + kw * 64;
      bm0.v = *(const v16bf*)&bp[kb];
      bm1.v = *(const v16bf*)&bp[32 + kb];
      acc0 = __builtin_amdgcn_wmma_f32_16x16x32_bf16(false, a0.v, false, bm0.v,
                                                     (short)0, acc0, false, false);
      acc1 = __builtin_amdgcn_wmma_f32_16x16x32_bf16(false, a1.v, false, bm1.v,
                                                     (short)0, acc1, false, false);
      ap += 64;
    }
  }
  const int w = w0 + row;
  const int h = h0 + hsub;
  if (w < CW) {
#pragma unroll
    for (int r = 0; r < 8; ++r) {
      int o = o0 + r + ((lane < 16) ? 0 : 8);
      if (o < CK)
        attn[((size_t)bh * CK + o) * CHW + h * CW + w] = acc0[r] + acc1[r] + dm_b[o];
    }
  }
}

// ---------------- softmax over 49 taps (+rel_bias), fp32 in place ----------------
__global__ void softmax_kernel(float* __restrict__ attn, const float* __restrict__ rel_bias) {
  int idx = blockIdx.x * blockDim.x + threadIdx.x;
  if (idx >= CBH * CHW) return;
  int bh = idx / CHW, hw = idx % CHW;
  int head = bh % CHEADS;
  size_t base = (size_t)bh * CK * CHW + hw;
  float a[CK];
  float mx = -1e30f;
  for (int k = 0; k < CK; ++k) {
    float t = attn[base + (size_t)k * CHW] + rel_bias[k * CHEADS + head];
    a[k] = t;
    mx = fmaxf(mx, t);
  }
  float s = 0.f;
  for (int k = 0; k < CK; ++k) { a[k] = __expf(a[k] - mx); s += a[k]; }
  float inv = 1.f / s;
  for (int k = 0; k < CK; ++k) attn[base + (size_t)k * CHW] = a[k] * inv;
}

// ---------------- attention * v neighborhood gather (batched matvec, VALU) ----------------
__global__ void av_kernel(const __bf16* __restrict__ vb, const float* __restrict__ attn,
                          __bf16* __restrict__ av) {
  int idx = blockIdx.x * blockDim.x + threadIdx.x;
  if (idx >= CBH * CHD * CHW) return;
  int hw = idx % CHW;
  int t  = idx / CHW;
  int c  = t % CHD;
  int bh = t / CHD;
  int h = hw / CW, w = hw % CW;
  const __bf16* vp = &vb[((size_t)bh * CHD + c) * CHW];
  const float*  ap = &attn[(size_t)bh * CK * CHW + hw];
  float acc = 0.f;
  for (int kh = 0; kh < CWIN; ++kh) {
    int gh = h + kh - CPAD;
    if (gh < 0 || gh >= CH) continue;
    for (int kw = 0; kw < CWIN; ++kw) {
      int gw = w + kw - CPAD;
      if (gw < 0 || gw >= CW) continue;
      acc += bf2f(vp[gh * CW + gw]) * ap[(size_t)(kh * CWIN + kw) * CHW];
    }
  }
  int b = bh / CHEADS, head = bh % CHEADS;
  av[((size_t)(b * CDIM + head * CHD + c)) * CHW + hw] = f2bf(acc);
}

// ---------------- host launch ----------------
extern "C" void kernel_launch(void* const* d_in, const int* in_sizes, int n_in,
                              void* d_out, int out_size, void* d_ws, size_t ws_size,
                              hipStream_t stream) {
  const float* x      = (const float*)d_in[0];
  const float* gamma  = (const float*)d_in[1];
  const float* beta   = (const float*)d_in[2];
  const float* qkv_w  = (const float*)d_in[3];
  const float* qkv_b  = (const float*)d_in[4];
  const float* dm_w   = (const float*)d_in[5];
  const float* dm_b   = (const float*)d_in[6];
  const float* rel_b  = (const float*)d_in[7];
  const float* proj_w = (const float*)d_in[8];
  const float* proj_b = (const float*)d_in[9];
  const float* c1_w   = (const float*)d_in[10];
  const float* c1_b   = (const float*)d_in[11];
  const float* c2_w   = (const float*)d_in[12];
  const float* c2_b   = (const float*)d_in[13];

  char* ws = (char*)d_ws;
  float*  w_scale = (float*)(ws + OFF_SCALE);
  float*  w_shift = (float*)(ws + OFF_SHIFT);
  __bf16* w_xn    = (__bf16*)(ws + OFF_XN);
  __bf16* w_wqkv  = (__bf16*)(ws + OFF_WQKV);
  __bf16* w_wproj = (__bf16*)(ws + OFF_WPROJ);
  __bf16* w_wc1   = (__bf16*)(ws + OFF_WC1);
  __bf16* w_wc2   = (__bf16*)(ws + OFF_WC2);
  __bf16* w_wdm   = (__bf16*)(ws + OFF_WDM);
  __bf16* w_qk    = (__bf16*)(ws + OFF_QK);
  __bf16* w_v     = (__bf16*)(ws + OFF_V);
  float*  w_att   = (float*)(ws + OFF_ATT);
  __bf16* w_av    = (__bf16*)(ws + OFF_AV);
  float*  w_x1f   = (float*)(ws + OFF_X1F);
  __bf16* w_x1b   = (__bf16*)(ws + OFF_X1B);
  __bf16* w_h     = (__bf16*)(ws + OFF_H);
  float*  out     = (float*)d_out;

  // 1) weights -> bf16
  cvt_bf16_kernel<<<(3*CDIM*CDIM + 255)/256, 256, 0, stream>>>(w_wqkv,  qkv_w,  3*CDIM*CDIM);
  cvt_bf16_kernel<<<(CDIM*CDIM   + 255)/256, 256, 0, stream>>>(w_wproj, proj_w, CDIM*CDIM);
  cvt_bf16_kernel<<<(4*CDIM*CDIM + 255)/256, 256, 0, stream>>>(w_wc1,   c1_w,   4*CDIM*CDIM);
  cvt_bf16_kernel<<<(4*CDIM*CDIM + 255)/256, 256, 0, stream>>>(w_wc2,   c2_w,   4*CDIM*CDIM);
  dmw_transpose_kernel<<<(CKP*CK*64 + 255)/256, 256, 0, stream>>>(w_wdm, dm_w);

  // 2) BN stats + normalize to bf16
  bn_stats_kernel<<<CDIM, 256, 0, stream>>>(x, gamma, beta, w_scale, w_shift);
  normalize_kernel<<<(CNPIX*CDIM + 255)/256, 256, 0, stream>>>(x, w_scale, w_shift, w_xn);

  const int nblocks = CNPIX / 32;   // 392

  // 3) QKV GEMM (576x192x12544) -> qk pixel-major + v channel-major
  gemm_wmma_kernel<0><<<dim3(nblocks, (3*CDIM)/64), 256, 0, stream>>>(
      w_wqkv, w_xn, qkv_b, nullptr, nullptr, w_qk, w_v, CDIM);

  // 4) attention logits via implicit-GEMM conv (+dm_b); 2 rows per block
  attn_conv_kernel<<<dim3(4, CH/2, CBH), 256, 0, stream>>>(w_qk, w_wdm, dm_b, w_att);

  // 5) softmax over taps (+rel_bias)
  softmax_kernel<<<(CBH*CHW + 255)/256, 256, 0, stream>>>(w_att, rel_b);

  // 6) attention-weighted v gather
  av_kernel<<<(CBH*CHD*CHW + 255)/256, 256, 0, stream>>>(w_v, w_att, w_av);

  // 7) proj + shortcut residual -> x1 (fp32 + bf16)
  gemm_wmma_kernel<1><<<dim3(nblocks, CDIM/64), 256, 0, stream>>>(
      w_wproj, w_av, proj_b, x, w_x1f, w_x1b, nullptr, CDIM);

  // 8) MLP up + exact GELU
  gemm_wmma_kernel<2><<<dim3(nblocks, (4*CDIM)/64), 256, 0, stream>>>(
      w_wc1, w_x1b, c1_b, nullptr, nullptr, w_h, nullptr, CDIM);

  // 9) MLP down + residual -> d_out
  gemm_wmma_kernel<3><<<dim3(nblocks, CDIM/64), 256, 0, stream>>>(
      w_wc2, w_h, c2_b, w_x1f, out, nullptr, nullptr, 4*CDIM);

  (void)in_sizes; (void)n_in; (void)out_size; (void)ws_size;
}
